// DecoderGRU_54374285968084
// MI455X (gfx1250) — matrix operand order repaired and checked by reading, not compiled, not run on hardware
//
#include <hip/hip_runtime.h>
#include <hip/hip_bf16.h>
#include <stdint.h>

#define Bn 8
#define Tn 64
#define Vn 30000
#define En 256
#define Hn 512
#define Mn 512
#define Un 32
#define Wn 512
#define Kn 1024          // H + M
#define NTILES (Vn / 16) // 1875, exact

typedef __attribute__((ext_vector_type(16))) __bf16 v16bf;
typedef __attribute__((ext_vector_type(8)))  __bf16 v8bf;
typedef __attribute__((ext_vector_type(8)))  float  v8f;

__device__ inline __bf16 f2bf(float f) {
    unsigned int u = __builtin_bit_cast(unsigned int, f);
    unsigned int r = u + 0x7FFFu + ((u >> 16) & 1u);   // round-to-nearest-even
    unsigned short h = (unsigned short)(r >> 16);
    return __builtin_bit_cast(__bf16, h);
}

// ---------------- one-time precompute kernels ----------------

__global__ void cvt_wo_kernel(const float* __restrict__ Wo, __bf16* __restrict__ Wob) {
    size_t N = (size_t)Vn * Kn;
    for (size_t i = (size_t)blockIdx.x * blockDim.x + threadIdx.x; i < N;
         i += (size_t)gridDim.x * blockDim.x)
        Wob[i] = f2bf(Wo[i]);
}

__global__ void zero_abf_kernel(__bf16* __restrict__ Abf) {
    int i = blockIdx.x * blockDim.x + threadIdx.x;
    if (i < 16 * Kn) Abf[i] = f2bf(0.0f);
}

__global__ void init_h_kernel(const float* __restrict__ u_output,
                              const int* __restrict__ u_len, float* __restrict__ h) {
    int b = blockIdx.x;
    int last = u_len[b] - 1;
    for (int k = threadIdx.x; k < Hn; k += blockDim.x)
        h[b * Hn + k] = u_output[(b * Un + last) * Mn + k];
}

// Au[b,u,h] = u_output[b,u,:] . Wu[h,:] + bu[h]
__global__ void pre_au_kernel(const float* __restrict__ u_output, const float* __restrict__ Wu,
                              const float* __restrict__ bu, float* __restrict__ Au) {
    int idx = blockIdx.x * blockDim.x + threadIdx.x;
    if (idx >= Bn * Un * Hn) return;
    int hh = idx % Hn;
    int bu_i = idx / Hn;
    const float* x = u_output + bu_i * Mn;
    const float* w = Wu + hh * Mn;
    float p = bu[hh];
#pragma unroll 4
    for (int m = 0; m < Mn; m++) p += x[m] * w[m];
    Au[idx] = p;
}

// Aw[b,w,h] = w_output[b,w,:] . Ww[h,:] + bw[h]  (LDS-tiled, 64x64 tiles)
__global__ __launch_bounds__(256) void pre_aw_kernel(const float* __restrict__ w_output,
                                                     const float* __restrict__ Ww,
                                                     const float* __restrict__ bw,
                                                     float* __restrict__ Aw) {
    __shared__ float sX[64][17];
    __shared__ float sW[64][17];
    int bid = blockIdx.x;
    int ht = bid % (Hn / 64);
    int wt = (bid / (Hn / 64)) % (Wn / 64);
    int b  = bid / ((Hn / 64) * (Wn / 64));
    int w0 = wt * 64, h0 = ht * 64;
    int tx = threadIdx.x & 15;   // h micro group
    int ty = threadIdx.x >> 4;   // w micro group
    float acc[4][4];
#pragma unroll
    for (int i = 0; i < 4; i++)
#pragma unroll
        for (int j = 0; j < 4; j++) acc[i][j] = 0.0f;

    for (int m0 = 0; m0 < Mn; m0 += 16) {
        __syncthreads();
        int base = threadIdx.x * 4;            // 1024 elems / 256 threads
        int row = base >> 4, col = base & 15;
#pragma unroll
        for (int j = 0; j < 4; j++) {
            sX[row][col + j] = w_output[(b * Wn + w0 + row) * Mn + m0 + col + j];
            sW[row][col + j] = Ww[(h0 + row) * Mn + m0 + col + j];
        }
        __syncthreads();
#pragma unroll
        for (int mm = 0; mm < 16; mm++) {
            float xv[4], wv[4];
#pragma unroll
            for (int i = 0; i < 4; i++) xv[i] = sX[ty * 4 + i][mm];
#pragma unroll
            for (int j = 0; j < 4; j++) wv[j] = sW[tx * 4 + j][mm];
#pragma unroll
            for (int i = 0; i < 4; i++)
#pragma unroll
                for (int j = 0; j < 4; j++) acc[i][j] += xv[i] * wv[j];
        }
    }
#pragma unroll
    for (int i = 0; i < 4; i++)
#pragma unroll
        for (int j = 0; j < 4; j++)
            Aw[(b * Wn + w0 + ty * 4 + i) * Hn + h0 + tx * 4 + j] =
                acc[i][j] + bw[h0 + tx * 4 + j];
}

// ---------------- per-step kernels ----------------

// One block per batch: GRU step + utterance/word attention + gamma gate + ctx.
__global__ __launch_bounds__(256) void step_kernel(
    int t, const int* __restrict__ target, const float* __restrict__ emb,
    const float* __restrict__ W_ih, const float* __restrict__ W_hh,
    const float* __restrict__ b_ih, const float* __restrict__ b_hh,
    const float* __restrict__ Wd, const float* __restrict__ Wy,
    const float* __restrict__ Ws, const float* __restrict__ bs,
    const float* __restrict__ w_output, const int* __restrict__ u_len,
    const int* __restrict__ w_len, const int* __restrict__ word2utt,
    const float* __restrict__ Au, const float* __restrict__ Aw,
    float* __restrict__ h, __bf16* __restrict__ Abf,
    const float* __restrict__ out_dec, float* __restrict__ out_suw,
    float* __restrict__ out_su) {
    int b = blockIdx.x;
    int tid = threadIdx.x;
    __shared__ float sy[En];
    __shared__ float sh[Hn];
    __shared__ float shn[Hn];
    __shared__ float ssw[Wn];
    __shared__ float ssuw[Wn];
    __shared__ float ssu[Un];
    __shared__ float sm[Un], sd[Un];
    __shared__ float red[256];
    __shared__ float sgamma;

    int yt = target[b * Tn + t];
    for (int e = tid; e < En; e += 256) sy[e] = emb[yt * En + e];
    for (int k = tid; k < Hn; k += 256) sh[k] = h[b * Hn + k];
    __syncthreads();

    // --- GRU step (fp32) ---
    for (int k = tid; k < Hn; k += 256) {
        float ir = b_ih[k], iz = b_ih[Hn + k], in = b_ih[2 * Hn + k];
        const float* wr = W_ih + k * En;
        const float* wz = W_ih + (Hn + k) * En;
        const float* wn = W_ih + (2 * Hn + k) * En;
#pragma unroll 4
        for (int e = 0; e < En; e++) {
            float yv = sy[e];
            ir += wr[e] * yv; iz += wz[e] * yv; in += wn[e] * yv;
        }
        float hr = b_hh[k], hz = b_hh[Hn + k], hn = b_hh[2 * Hn + k];
        const float* vr = W_hh + k * Hn;
        const float* vz = W_hh + (Hn + k) * Hn;
        const float* vn = W_hh + (2 * Hn + k) * Hn;
#pragma unroll 4
        for (int e = 0; e < Hn; e++) {
            float hv = sh[e];
            hr += vr[e] * hv; hz += vz[e] * hv; hn += vn[e] * hv;
        }
        float r = 1.0f / (1.0f + __expf(-(ir + hr)));
        float z = 1.0f / (1.0f + __expf(-(iz + hz)));
        float n = tanhf(in + r * hn);
        shn[k] = (1.0f - z) * n + z * sh[k];
    }
    __syncthreads();
    for (int k = tid; k < Hn; k += 256) {
        h[b * Hn + k] = shn[k];
        Abf[b * Kn + Mn + k] = f2bf(shn[k]);   // A row, h part
    }

    // --- su = softmax_masked(h_new . Au) ---
    {
        int u = tid >> 3, l = tid & 7;
        const float* au = Au + (b * Un + u) * Hn;
        float p = 0.0f;
        for (int k = l; k < Hn; k += 8) p += shn[k] * au[k];
        red[tid] = p;
    }
    __syncthreads();
    if (tid < Un) {
        float s = 0.0f;
#pragma unroll
        for (int j = 0; j < 8; j++) s += red[tid * 8 + j];
        ssu[tid] = s;
    }
    __syncthreads();
    if (tid == 0) {
        int ul = u_len[b];
        float m = -1e30f;
        for (int u = 0; u < Un; u++) if (u < ul && ssu[u] > m) m = ssu[u];
        float s = 0.0f;
        for (int u = 0; u < Un; u++) {
            float e = (u < ul) ? __expf(ssu[u] - m) : 0.0f;
            ssu[u] = e; s += e;
        }
        for (int u = 0; u < Un; u++) ssu[u] /= s;
    }
    __syncthreads();

    // --- gamma = sigmoid(d_prev.Wd + y.Wy + h_new.Ws + bs) ---
    {
        float p = 0.0f;
        if (t > 0) {
            const float* dprev = out_dec + ((size_t)b * Tn + (t - 1)) * Vn;
            for (int v = tid; v < Vn; v += 256) p += dprev[v] * Wd[v];
        }
        for (int e = tid; e < En; e += 256) p += sy[e] * Wy[e];
        for (int k = tid; k < Hn; k += 256) p += shn[k] * Ws[k];
        red[tid] = p;
        __syncthreads();
        for (int s = 128; s > 0; s >>= 1) {
            if (tid < s) red[tid] += red[tid + s];
            __syncthreads();
        }
        if (tid == 0) sgamma = 1.0f / (1.0f + __expf(-(red[0] + bs[0])));
        __syncthreads();
    }

    // --- mix with eu_prev, renormalize, emit scores_u ---
    if (tid == 0) {
        float g = sgamma, s = 0.0f;
        for (int u = 0; u < Un; u++) {
            float eu = (t > 0) ? out_su[(b * Tn + (t - 1)) * Un + u] : 0.0f;
            float v = (1.0f - g) * ssu[u] + g * eu;
            ssu[u] = v; s += v;
        }
        for (int u = 0; u < Un; u++) {
            ssu[u] /= s;
            out_su[(b * Tn + t) * Un + u] = ssu[u];
        }
    }
    __syncthreads();

    // --- sw = h_new . Aw ---
    for (int w = tid; w < Wn; w += 256) {
        const float* aw = Aw + (b * Wn + w) * Hn;
        float p = 0.0f;
#pragma unroll 4
        for (int k = 0; k < Hn; k++) p += shn[k] * aw[k];
        ssw[w] = p;
    }
    __syncthreads();
    // per-utterance segment softmax stats
    if (tid < Un) {
        int wl = w_len[b];
        float m = -1e30f;
        for (int w = 0; w < Wn; w++)
            if (w < wl && word2utt[b * Wn + w] == tid && ssw[w] > m) m = ssw[w];
        float s = 0.0f;
        for (int w = 0; w < Wn; w++)
            if (w < wl && word2utt[b * Wn + w] == tid) s += __expf(ssw[w] - m);
        sm[tid] = m; sd[tid] = s;
    }
    __syncthreads();
    for (int w = tid; w < Wn; w += 256) {
        int seg = word2utt[b * Wn + w];
        float pw = (w < w_len[b] && sd[seg] > 0.0f) ? __expf(ssw[w] - sm[seg]) / sd[seg] : 0.0f;
        float v = ssu[seg] * pw;
        ssuw[w] = v;
        out_suw[(b * Tn + t) * Wn + w] = v;
    }
    __syncthreads();
    // --- ctx = suw @ w_output, emit bf16 A row (ctx part) ---
    for (int m = tid; m < Mn; m += 256) {
        float p = 0.0f;
#pragma unroll 4
        for (int w = 0; w < Wn; w++) p += ssuw[w] * w_output[(b * Wn + w) * Mn + m];
        Abf[b * Kn + m] = f2bf(p);
    }
}

// logits[8,30000] = A[16,1024](bf16, rows 8..15 zero) @ Wo_bf16^T + bo
// One wave per 16-column vocab tile, v_wmma_f32_16x16x32_bf16 over K=1024.
__global__ __launch_bounds__(128) void vocab_gemm_kernel(const __bf16* __restrict__ Abf,
                                                         const __bf16* __restrict__ Wob,
                                                         const float* __restrict__ bo,
                                                         float* __restrict__ logits) {
    int wave = threadIdx.x >> 5;
    int lane = threadIdx.x & 31;
    int tile = blockIdx.x * 4 + wave;     // wave-uniform
    if (tile >= NTILES) return;            // whole wave exits together
    int vbase = tile * 16;
    int n  = lane & 15;
    int hi = lane >> 4;                    // 0: lanes 0-15, 1: lanes 16-31

    v8f acc = {};
    const __bf16* arow = Abf + n * Kn;                          // A row m = lane&15
    const __bf16* brow = Wob + (size_t)(vbase + n) * Kn;        // B col n = lane&15
#pragma unroll 4
    for (int kk = 0; kk < Kn; kk += 32) {
        // A 16x32 bf16: lane m holds K {hi*8..hi*8+7} and {16+hi*8..16+hi*8+7}
        v8bf a0 = *(const v8bf*)(arow + kk + (hi << 3));
        v8bf a1 = *(const v8bf*)(arow + kk + 16 + (hi << 3));
        // B 32x16 bf16: lane n(+16*hi) holds K {hi*8..hi*8+7} and {16+hi*8..16+hi*8+7}
        v8bf b0 = *(const v8bf*)(brow + kk + (hi << 3));
        v8bf b1 = *(const v8bf*)(brow + kk + 16 + (hi << 3));
        v16bf av, bv;
#pragma unroll
        for (int i = 0; i < 8; i++) {
            av[i] = a0[i]; av[i + 8] = a1[i];
            bv[i] = b0[i]; bv[i + 8] = b1[i];
        }
        acc = __builtin_amdgcn_wmma_f32_16x16x32_bf16(false, av, false, bv,
                                                      (short)0, acc, false, false);
    }
    // D layout: lanes 0-15 hold rows M=0..7 in acc[0..7], col N = lane
    if (hi == 0) {
        int v = vbase + n;
        float bias = bo[v];
#pragma unroll
        for (int i = 0; i < 8; i++)
            logits[(size_t)i * Vn + v] = acc[i] + bias;
    }
}

// dt = log_softmax(logits) per batch, written to d_out
__global__ __launch_bounds__(256) void lsm_kernel(int t, const float* __restrict__ logits,
                                                  float* __restrict__ out_dec) {
    int b = blockIdx.x, tid = threadIdx.x;
    __shared__ float red[256];
    const float* L = logits + (size_t)b * Vn;
    float m = -1e30f;
    for (int v = tid; v < Vn; v += 256) m = fmaxf(m, L[v]);
    red[tid] = m;
    __syncthreads();
    for (int s = 128; s > 0; s >>= 1) {
        if (tid < s) red[tid] = fmaxf(red[tid], red[tid + s]);
        __syncthreads();
    }
    m = red[0];
    __syncthreads();
    float s = 0.0f;
    for (int v = tid; v < Vn; v += 256) s += __expf(L[v] - m);
    red[tid] = s;
    __syncthreads();
    for (int st = 128; st > 0; st >>= 1) {
        if (tid < st) red[tid] += red[tid + st];
        __syncthreads();
    }
    float lse = m + __logf(red[0]);
    float* D = out_dec + ((size_t)b * Tn + t) * Vn;
    for (int v = tid; v < Vn; v += 256) D[v] = L[v] - lse;
}

// ---------------- launch ----------------

extern "C" void kernel_launch(void* const* d_in, const int* in_sizes, int n_in,
                              void* d_out, int out_size, void* d_ws, size_t ws_size,
                              hipStream_t stream) {
    (void)in_sizes; (void)n_in; (void)out_size; (void)ws_size;
    const int*   target   = (const int*)d_in[0];
    const float* u_output = (const float*)d_in[1];
    const float* w_output = (const float*)d_in[2];
    const int*   u_len    = (const int*)d_in[3];
    const int*   w_len    = (const int*)d_in[4];
    const int*   word2utt = (const int*)d_in[5];
    const float* emb      = (const float*)d_in[6];
    const float* W_ih     = (const float*)d_in[7];
    const float* W_hh     = (const float*)d_in[8];
    const float* b_ih     = (const float*)d_in[9];
    const float* b_hh     = (const float*)d_in[10];
    const float* Wu       = (const float*)d_in[11];
    const float* bu       = (const float*)d_in[12];
    const float* Ww       = (const float*)d_in[13];
    const float* bw       = (const float*)d_in[14];
    const float* Wo       = (const float*)d_in[15];
    const float* bo       = (const float*)d_in[16];
    const float* Wd       = (const float*)d_in[17];
    const float* Wy       = (const float*)d_in[18];
    const float* Ws       = (const float*)d_in[19];
    const float* bs       = (const float*)d_in[20];

    // workspace layout (256B aligned)
    char* base = (char*)d_ws;
    size_t off = 0;
    auto take = [&](size_t bytes) {
        size_t o = off;
        off = (off + bytes + 255) & ~(size_t)255;
        return o;
    };
    __bf16* Wob    = (__bf16*)(base + take((size_t)Vn * Kn * 2));
    float*  Au     = (float*)(base + take((size_t)Bn * Un * Hn * 4));
    float*  Aw     = (float*)(base + take((size_t)Bn * Wn * Hn * 4));
    float*  hstate = (float*)(base + take((size_t)Bn * Hn * 4));
    __bf16* Abf    = (__bf16*)(base + take((size_t)16 * Kn * 2));
    float*  logits = (float*)(base + take((size_t)Bn * Vn * 4));

    float* out = (float*)d_out;
    float* out_dec = out;
    float* out_suw = out + (size_t)Bn * Tn * Vn;
    float* out_su  = out_suw + (size_t)Bn * Tn * Wn;

    // one-time precompute
    cvt_wo_kernel<<<2048, 256, 0, stream>>>(Wo, Wob);
    zero_abf_kernel<<<(16 * Kn + 255) / 256, 256, 0, stream>>>(Abf);
    init_h_kernel<<<Bn, 256, 0, stream>>>(u_output, u_len, hstate);
    pre_au_kernel<<<(Bn * Un * Hn + 255) / 256, 256, 0, stream>>>(u_output, Wu, bu, Au);
    pre_aw_kernel<<<Bn * (Wn / 64) * (Hn / 64), 256, 0, stream>>>(w_output, Ww, bw, Aw);

    // sequential decode
    for (int t = 0; t < Tn; t++) {
        step_kernel<<<Bn, 256, 0, stream>>>(t, target, emb, W_ih, W_hh, b_ih, b_hh,
                                            Wd, Wy, Ws, bs, w_output, u_len, w_len,
                                            word2utt, Au, Aw, hstate, Abf,
                                            out_dec, out_suw, out_su);
        vocab_gemm_kernel<<<(NTILES + 3) / 4, 128, 0, stream>>>(Abf, Wob, bo, logits);
        lsm_kernel<<<Bn, 256, 0, stream>>>(t, logits, out_dec);
    }
}
